// GE2ELoss_23948737642964
// MI455X (gfx1250) — matrix operand order repaired and checked by reading, not compiled, not run on hardware
//
#include <hip/hip_runtime.h>
#include <hip/hip_bf16.h>

// ---------------------------------------------------------------------------
// GE2E loss, fused for gfx1250 (MI455X).
//   N_SPK=2048, M_UTT=10, D_EMB=128  -> rows R = 20480 query vectors.
//   Dominant cost: [R x 128] x [128 x 2048] cosine-sim GEMM (10.7 GFLOP).
//   bf16 hi/lo split-GEMM (AhiBhi + AhiBlo + AloBhi, f32 acc, ~2^-16 rel err)
//   on v_wmma_f32_16x16x32_bf16; flash-style online logsumexp fused in
//   registers (168 MB sims matrix never materialized).
//   B tiles are double-buffered in LDS: the whole block cooperatively stages
//   tile kt+1 from L2 while the WMMA chain consumes tile kt from LDS
//   (bank-conflict-free via 272B column padding). A tiles stay in VGPRs.
// ---------------------------------------------------------------------------

#define N_SPK 2048
#define M_UTT 10
#define D_EMB 128
#define N_ROWS (N_SPK * M_UTT)           // 20480
#define ROWS_PER_BLOCK 64                // 4 waves x 16 rows
#define N_BLOCKS2 (N_ROWS / ROWS_PER_BLOCK)  // 320
#define N_KTILES (N_SPK / 16)            // 128
#define GE2E_EPS 1e-12f

// LDS B-tile geometry: [hi/lo][16 cols][128 dims bf16], column padded to 272B
#define COL_BYTES   272                  // 256B data + 16B pad (17 uint4)
#define COL_U4      17
#define HILO_U4     (16 * COL_U4)        // 272 uint4 per hi/lo block
#define BUF_U4      (2 * HILO_U4)        // 544 uint4 per buffer (8704 B)
#define BUF_BYTES   (BUF_U4 * 16)

typedef __attribute__((ext_vector_type(16))) __bf16 v16bf;
typedef __attribute__((ext_vector_type(8)))  __bf16 v8bf;
typedef __attribute__((ext_vector_type(8)))  float  v8f;

// ---- bf16 helpers (RNE, via bit ops so rounding is explicit) --------------
__device__ __forceinline__ unsigned short f32_to_bf16_rne(float f) {
    unsigned int u = __float_as_uint(f);
    unsigned int r = u + 0x7FFFu + ((u >> 16) & 1u);
    return (unsigned short)(r >> 16);
}
__device__ __forceinline__ float bf16bits_to_f32(unsigned short h) {
    return __uint_as_float(((unsigned int)h) << 16);
}

__device__ __forceinline__ v16bf cat8(v8bf a, v8bf b) {
    v16bf r;
#pragma unroll
    for (int i = 0; i < 8; ++i) { r[i] = a[i]; r[8 + i] = b[i]; }
    return r;
}

// ---- block-wide sum over 128 threads (4 wave32) ---------------------------
__device__ __forceinline__ float blockReduceSum128(float v, float* red) {
#pragma unroll
    for (int off = 16; off > 0; off >>= 1) v += __shfl_xor(v, off, 32);
    int wave = threadIdx.x >> 5;
    if ((threadIdx.x & 31) == 0) red[wave] = v;
    __syncthreads();
    float r = red[0] + red[1] + red[2] + red[3];
    __syncthreads();
    return r;
}

// ===========================================================================
// Kernel 1: per speaker -- centroids, normalizations, bf16 hi/lo split,
//           self (excluded-centroid) similarities.
//   grid = N_SPK blocks, 128 threads (one thread per embedding dim).
// ===========================================================================
__global__ void __launch_bounds__(128)
ge2e_prep(const float* __restrict__ dvecs, const float* __restrict__ wp,
          const float* __restrict__ bp,
          unsigned short* __restrict__ Dhi, unsigned short* __restrict__ Dlo,
          unsigned short* __restrict__ CIhi, unsigned short* __restrict__ CIlo,
          float* __restrict__ pos) {
    __shared__ float red[4];
    const int n = blockIdx.x;
    const int t = threadIdx.x;                      // embedding dim
    const float* base = dvecs + (size_t)n * M_UTT * D_EMB;

    float x[M_UTT];
    float c = 0.f;
#pragma unroll
    for (int m = 0; m < M_UTT; ++m) { x[m] = base[m * D_EMB + t]; c += x[m]; }
    c *= (1.0f / M_UTT);                            // inclusive centroid (dim t)

    // normalized inclusive centroid -> CIhi/CIlo (bf16 split)
    float nc2 = blockReduceSum128(c * c, red);
    float ci  = c / fmaxf(sqrtf(nc2), GE2E_EPS);
    unsigned short hb = f32_to_bf16_rne(ci);
    unsigned short lb = f32_to_bf16_rne(ci - bf16bits_to_f32(hb));
    CIhi[(size_t)n * D_EMB + t] = hb;
    CIlo[(size_t)n * D_EMB + t] = lb;

    const float wAbs = fabsf(wp[0]);
    const float bias = bp[0];

#pragma unroll
    for (int m = 0; m < M_UTT; ++m) {
        float e = (c * (float)M_UTT - x[m]) * (1.0f / (float)(M_UTT - 1));
        float sx  = blockReduceSum128(x[m] * x[m], red);   // ||x||^2
        float se  = blockReduceSum128(e * e, red);         // ||c_excl||^2
        float sxe = blockReduceSum128(x[m] * e, red);      // x . c_excl

        float nx = fmaxf(sqrtf(sx), GE2E_EPS);
        float ne = fmaxf(sqrtf(se), GE2E_EPS);
        float dn = x[m] / nx;                              // normalized utterance

        size_t row = (size_t)n * M_UTT + m;
        unsigned short dh = f32_to_bf16_rne(dn);
        unsigned short dl = f32_to_bf16_rne(dn - bf16bits_to_f32(dh));
        Dhi[row * D_EMB + t] = dh;
        Dlo[row * D_EMB + t] = dl;

        if (t == 0) pos[row] = wAbs * (sxe / (nx * ne)) + bias;  // diag logit
    }
}

// ===========================================================================
// Kernel 2: WMMA split-GEMM + fused online logsumexp, LDS-pipelined B.
//   grid = 320 blocks x 128 threads. Wave w owns rows [blk*64+w*16, +16);
//   its 16x128 A tile (hi+lo) stays in VGPRs for the whole column sweep.
// ===========================================================================
__global__ void __launch_bounds__(128)
ge2e_lse(const unsigned short* __restrict__ Dhi,
         const unsigned short* __restrict__ Dlo,
         const unsigned short* __restrict__ CIhi,
         const unsigned short* __restrict__ CIlo,
         const float* __restrict__ pos,
         const float* __restrict__ wp, const float* __restrict__ bp,
         float* __restrict__ partials) {
    __shared__ uint4 bbuf[2 * BUF_U4];   // double-buffered B tile, 17408 B
    __shared__ float wpart[8];

    const int lane    = threadIdx.x & 31;
    const int wave    = threadIdx.x >> 5;
    const int halfSel = lane >> 4;          // 0: lanes 0-15, 1: lanes 16-31
    const int laneMod = lane & 15;
    const int rowBase = blockIdx.x * ROWS_PER_BLOCK + wave * 16;

    const float wAbs = fabsf(wp[0]);
    const float bias = bp[0];

    // ---- cooperative B staging: thread -> (2 cols) x (16B seg), hi & lo ---
    const int colA = threadIdx.x >> 4;                 // 0..7
    const int seg  = threadIdx.x & 15;                 // 16B segment
    const unsigned short* gHs = CIhi + (size_t)colA * D_EMB + seg * 8;
    const unsigned short* gLs = CIlo + (size_t)colA * D_EMB + seg * 8;
    const size_t tileStride = (size_t)16 * D_EMB;      // elements per k-tile

    // ---- loop-invariant A tile (ISA 16-bit A 16x32 layout) ----------------
    const unsigned short* arowh = Dhi + (size_t)(rowBase + laneMod) * D_EMB;
    const unsigned short* arowl = Dlo + (size_t)(rowBase + laneMod) * D_EMB;
    v16bf ahi[4], alo[4];
#pragma unroll
    for (int c = 0; c < 4; ++c) {
        int o0 = 32 * c + 8 * halfSel;
        int o1 = o0 + 16;
        ahi[c] = cat8(*(const v8bf*)(arowh + o0), *(const v8bf*)(arowh + o1));
        alo[c] = cat8(*(const v8bf*)(arowl + o0), *(const v8bf*)(arowl + o1));
    }

    // ---- per-lane row metadata (C/D layout: vgpr r -> M = r + 8*halfSel) --
    int   jr[8];
    float posr[8];
#pragma unroll
    for (int r = 0; r < 8; ++r) {
        int row  = rowBase + r + 8 * halfSel;
        jr[r]    = row / M_UTT;             // speaker index of this row
        posr[r]  = pos[row];                // |w|*self_sim + b
    }

    // ---- online logsumexp state -------------------------------------------
    float mr[8], sr[8];
#pragma unroll
    for (int r = 0; r < 8; ++r) { mr[r] = -__builtin_inff(); sr[r] = 0.f; }

    // ---- per-lane LDS read base (B 32x16 layout; 272B padded columns) -----
    const char* ldsRd = (const char*)bbuf + laneMod * COL_BYTES + 16 * halfSel;

    // ---- prime buffer 0 ----------------------------------------------------
    uint4 stg[4];
    stg[0] = *(const uint4*)(gHs);
    stg[1] = *(const uint4*)(gHs + 8 * D_EMB);
    stg[2] = *(const uint4*)(gLs);
    stg[3] = *(const uint4*)(gLs + 8 * D_EMB);
    {
        uint4* d = bbuf;
        d[colA * COL_U4 + seg]                 = stg[0];
        d[(colA + 8) * COL_U4 + seg]           = stg[1];
        d[HILO_U4 + colA * COL_U4 + seg]       = stg[2];
        d[HILO_U4 + (colA + 8) * COL_U4 + seg] = stg[3];
    }
    __syncthreads();

    for (int kt = 0; kt < N_KTILES; ++kt) {
        const int cur = kt & 1;

        // stage tile kt+1 from global: loads fly while WMMAs below execute
        if (kt + 1 < N_KTILES) {
            const unsigned short* h = gHs + (size_t)(kt + 1) * tileStride;
            const unsigned short* l = gLs + (size_t)(kt + 1) * tileStride;
            __builtin_prefetch(h + tileStride, 0, 1);   // kt+2 -> global_prefetch
            stg[0] = *(const uint4*)(h);
            stg[1] = *(const uint4*)(h + 8 * D_EMB);
            stg[2] = *(const uint4*)(l);
            stg[3] = *(const uint4*)(l + 8 * D_EMB);
        }

        // ---- 12 WMMAs on LDS buffer `cur` ---------------------------------
        const char* bH = ldsRd + cur * BUF_BYTES;          // hi block
        const char* bL = bH + HILO_U4 * 16;                // lo block (+4352B)
        v8f acc = {};
#pragma unroll
        for (int c = 0; c < 4; ++c) {
            const v8bf* ph = (const v8bf*)(bH + 64 * c);
            const v8bf* pl = (const v8bf*)(bL + 64 * c);
            v16bf bhi = cat8(ph[0], ph[2]);
            v16bf blo = cat8(pl[0], pl[2]);
            // split product: AhiBhi + AhiBlo + AloBhi  (f32 accumulate)
            acc = __builtin_amdgcn_wmma_f32_16x16x32_bf16(
                      false, ahi[c], false, bhi, (short)0, acc, false, false);
            acc = __builtin_amdgcn_wmma_f32_16x16x32_bf16(
                      false, ahi[c], false, blo, (short)0, acc, false, false);
            acc = __builtin_amdgcn_wmma_f32_16x16x32_bf16(
                      false, alo[c], false, bhi, (short)0, acc, false, false);
        }

        // ---- fold tile into online logsumexp; diag -> self logit ----------
        const int colIdx = kt * 16 + laneMod;
#pragma unroll
        for (int r = 0; r < 8; ++r) {
            float v = wAbs * acc[r] + bias;
            if (colIdx == jr[r]) v = posr[r];
            float nm = fmaxf(mr[r], v);
            sr[r] = sr[r] * __expf(mr[r] - nm) + __expf(v - nm);
            mr[r] = nm;
        }

        // ---- commit staged tile into the other LDS buffer -----------------
        if (kt + 1 < N_KTILES) {
            uint4* d = bbuf + (cur ^ 1) * BUF_U4;
            d[colA * COL_U4 + seg]                 = stg[0];
            d[(colA + 8) * COL_U4 + seg]           = stg[1];
            d[HILO_U4 + colA * COL_U4 + seg]       = stg[2];
            d[HILO_U4 + (colA + 8) * COL_U4 + seg] = stg[3];
        }
        __syncthreads();
    }

    // ---- combine the 16 lanes of each half-wave (columns are split) -------
#pragma unroll
    for (int r = 0; r < 8; ++r) {
#pragma unroll
        for (int off = 1; off < 16; off <<= 1) {
            float om = __shfl_xor(mr[r], off, 32);
            float os = __shfl_xor(sr[r], off, 32);
            float nm = fmaxf(mr[r], om);
            sr[r] = sr[r] * __expf(mr[r] - nm) + os * __expf(om - nm);
            mr[r] = nm;
        }
    }

    // per-row loss term: logsumexp - pos ; lane 0 of each 16-lane group owns it
    float part = 0.f;
    if (laneMod == 0) {
#pragma unroll
        for (int r = 0; r < 8; ++r) part += (__logf(sr[r]) + mr[r]) - posr[r];
        wpart[wave * 2 + halfSel] = part;
    }
    __syncthreads();
    if (threadIdx.x == 0) {
        float s = 0.f;
#pragma unroll
        for (int i = 0; i < 8; ++i) s += wpart[i];
        partials[blockIdx.x] = s;
    }
}

// ===========================================================================
// Kernel 3: deterministic final reduction of 320 partials -> mean loss.
// ===========================================================================
__global__ void __launch_bounds__(128)
ge2e_final(const float* __restrict__ partials, int n, float* __restrict__ out) {
    __shared__ float red[4];
    float s = 0.f;
    for (int i = threadIdx.x; i < n; i += 128) s += partials[i];
    s = blockReduceSum128(s, red);
    if (threadIdx.x == 0) out[0] = s / (float)N_ROWS;
}

// ===========================================================================
extern "C" void kernel_launch(void* const* d_in, const int* in_sizes, int n_in,
                              void* d_out, int out_size, void* d_ws, size_t ws_size,
                              hipStream_t stream) {
    (void)in_sizes; (void)n_in; (void)out_size; (void)ws_size;
    const float* dvecs = (const float*)d_in[0];
    const float* w     = (const float*)d_in[1];
    const float* b     = (const float*)d_in[2];
    float* out         = (float*)d_out;

    // workspace layout (256B aligned regions)
    char* ws = (char*)d_ws;
    size_t off = 0;
    auto take = [&](size_t bytes) -> char* {
        char* p = ws + off;
        off = (off + bytes + 255) & ~(size_t)255;
        return p;
    };
    unsigned short* Dhi  = (unsigned short*)take((size_t)N_ROWS * D_EMB * 2);
    unsigned short* Dlo  = (unsigned short*)take((size_t)N_ROWS * D_EMB * 2);
    unsigned short* CIhi = (unsigned short*)take((size_t)N_SPK * D_EMB * 2);
    unsigned short* CIlo = (unsigned short*)take((size_t)N_SPK * D_EMB * 2);
    float* pos           = (float*)take((size_t)N_ROWS * 4);
    float* partials      = (float*)take((size_t)N_BLOCKS2 * 4);

    ge2e_prep<<<N_SPK, 128, 0, stream>>>(dvecs, w, b, Dhi, Dlo, CIhi, CIlo, pos);
    ge2e_lse<<<N_BLOCKS2, 128, 0, stream>>>(Dhi, Dlo, CIhi, CIlo, pos, w, b, partials);
    ge2e_final<<<1, 128, 0, stream>>>(partials, N_BLOCKS2, out);
}
